// StructureAwareGraphAggr_42649025249986
// MI455X (gfx1250) — matrix operand order repaired and checked by reading, not compiled, not run on hardware
//
#include <hip/hip_runtime.h>
#include <cstdint>
#include <cmath>

// ---------------------------------------------------------------------------
// StructureAwareGraphAggr for MI455X (gfx1250, wave32, WMMA)
//
// Math refactor: the (Q,K,K) softmax and (Q,K,D) tensors are never
// materialized; dense algebra runs as bf16 WMMA GEMMs (f32 accumulate); the
// 268 MB rel_k/rel_v streams are b128-vectorized coalesced f32 reads fused
// with their consumers (memory-bound floor ~11.5us @ 23.3 TB/s).
// ---------------------------------------------------------------------------

#define DD   256
#define QN   256
#define KN   512
#define SCL  0.0625f   // 1/sqrt(256)

typedef __attribute__((ext_vector_type(16))) __bf16 v16bf;
typedef __attribute__((ext_vector_type(8)))  __bf16 v8bf;
typedef __attribute__((ext_vector_type(8)))  float  v8f;

__device__ __forceinline__ float sigf(float x) { return 1.0f / (1.0f + expf(-x)); }

// block-wide sum over 256 threads; returns result to all threads
__device__ __forceinline__ float block_reduce_sum(float v, float* red) {
    int t = threadIdx.x;
    red[t] = v;
    __syncthreads();
    #pragma unroll
    for (int s = 128; s > 0; s >>= 1) {
        if (t < s) red[t] += red[t + s];
        __syncthreads();
    }
    float r = red[0];
    __syncthreads();
    return r;
}

// ---------------- WMMA tile helpers (bf16 in, f32 acc) ----------------------
// A operand: 16x32 (MxK). Lane l: row M = l&15; VGPR0..3 -> K = off+{0..7},
// VGPR4..7 -> K = 16+off+{0..7}, off = (l<16?0:8). Two contiguous 16B loads.
__device__ __forceinline__ v16bf wmma_load_a(const __bf16* base, int ld,
                                             int row0, int col0, int lane) {
    int m   = lane & 15;
    int off = (lane & 16) ? 8 : 0;
    const __bf16* p = base + (size_t)(row0 + m) * ld + col0 + off;
    v8bf lo = *reinterpret_cast<const v8bf*>(p);        // K = off .. off+7
    v8bf hi = *reinterpret_cast<const v8bf*>(p + 16);   // K = 16+off .. 16+off+7
    v16bf a;
    #pragma unroll
    for (int i = 0; i < 8; ++i) { a[i] = lo[i]; a[8 + i] = hi[i]; }
    return a;
}

// B operand: 32x16 (KxN), element B[kk,n] = src[(n0+n)*ld + k0+kk].
// Lane l: col N = l&15, K = kb+{0..7}, 16+kb+{0..7}, kb = (l<16?0:8).
__device__ __forceinline__ v16bf wmma_load_b(const __bf16* base, int ld,
                                             int n0, int k0, int lane) {
    int n  = lane & 15;
    int kb = (lane & 16) ? 8 : 0;
    const __bf16* p = base + (size_t)(n0 + n) * ld + k0 + kb;
    v8bf lo = *reinterpret_cast<const v8bf*>(p);
    v8bf hi = *reinterpret_cast<const v8bf*>(p + 16);
    v16bf b;
    #pragma unroll
    for (int i = 0; i < 8; ++i) { b[i] = lo[i]; b[8 + i] = hi[i]; }
    return b;
}

// C/D: 16x16 f32, VGPR r: lanes 0-15 -> M=r, lanes 16-31 -> M=8+r; N = lane&15
__device__ __forceinline__ void wmma_store_c(float* base, int ld,
                                             int row0, int col0, int lane, v8f c) {
    int n  = lane & 15;
    int mb = (lane & 16) ? 8 : 0;
    float* p = base + (size_t)(row0 + mb) * ld + col0 + n;
    #pragma unroll
    for (int r = 0; r < 8; ++r) p[(size_t)r * ld] = c[r];
}

__device__ __forceinline__ v8f wmma_bf16(v16bf a, v16bf b, v8f c) {
    return __builtin_amdgcn_wmma_f32_16x16x32_bf16(false, a, false, b,
                                                   (short)0, c, false, false);
}

// ---------------------------------------------------------------------------
// K0: q_glob_enc = mean(q_enc); gq = qge@Wg+bg; g1 = qge@Wv0+bv0   (1 block)
__global__ void k_prep0(const float* __restrict__ q_enc,
                        const float* __restrict__ Wg,  const float* __restrict__ bg,
                        const float* __restrict__ Wv0, const float* __restrict__ bv0,
                        float* __restrict__ gq, float* __restrict__ g1) {
    __shared__ float sm[DD];
    int d = threadIdx.x;
    float s = 0.0f;
    for (int q = 0; q < QN; ++q) s += q_enc[(size_t)q * DD + d];
    s *= (1.0f / (float)QN);
    sm[d] = s;
    __syncthreads();
    float a = bg[d], b = bv0[d];
    for (int i = 0; i < DD; ++i) {
        float x = sm[i];
        a = fmaf(x, Wg[(size_t)i * DD + d], a);
        b = fmaf(x, Wv0[(size_t)i * DD + d], b);
    }
    gq[d] = a;
    g1[d] = b;
}

// K1: per-key row k: k_s gate, k1 = k_enc@Wv1+bv1, k_enc2 blend, s1/s2 dots.
// Writes k_enc2 (f32, row-major) and k2bT (bf16, transposed DxK for WMMA-B).
__global__ void k_kenc2(const float* __restrict__ k_enc,
                        const float* __restrict__ gq, const float* __restrict__ g1,
                        const float* __restrict__ Wv1, const float* __restrict__ bv1,
                        const float* __restrict__ Wlg1, const float* __restrict__ Wlg2,
                        float* __restrict__ k_enc2, __bf16* __restrict__ k2bT,
                        float* __restrict__ s1, float* __restrict__ s2) {
    __shared__ float krow[DD];
    __shared__ float red[256];
    int k = blockIdx.x, d = threadIdx.x;
    float ke = k_enc[(size_t)k * DD + d];
    krow[d] = ke;
    float dot = block_reduce_sum(gq[d] * ke, red);   // sync covers krow visibility
    float ks  = sigf(dot * SCL);
    float k1  = bv1[d];
    for (int i = 0; i < DD; ++i) k1 = fmaf(krow[i], Wv1[(size_t)i * DD + d], k1);
    float ke2 = (1.0f - ks) * k1 + ks * g1[d];
    k_enc2[(size_t)k * DD + d] = ke2;
    k2bT[(size_t)d * KN + k]   = (__bf16)ke2;
    float sv1 = block_reduce_sum(ke2 * Wlg1[d], red);
    float sv2 = block_reduce_sum(ke2 * Wlg2[d], red);
    if (d == 0) { s1[k] = sv1; s2[k] = sv2; }
}

// K2: q_glob = q_enc@Wgq+bgq, q_loc = q_enc@Wlq+blq (block per q row)
__global__ void k_qlin(const float* __restrict__ q_enc,
                       const float* __restrict__ Wgq, const float* __restrict__ bgq,
                       const float* __restrict__ Wlq, const float* __restrict__ blq,
                       float* __restrict__ q_glob, float* __restrict__ q_loc) {
    __shared__ float qrow[DD];
    int q = blockIdx.x, d = threadIdx.x;
    qrow[d] = q_enc[(size_t)q * DD + d];
    __syncthreads();
    float a = bgq[d], b = blq[d];
    for (int i = 0; i < DD; ++i) {
        float x = qrow[i];
        a = fmaf(x, Wgq[(size_t)i * DD + d], a);
        b = fmaf(x, Wlq[(size_t)i * DD + d], b);
    }
    q_glob[(size_t)q * DD + d] = a;
    q_loc [(size_t)q * DD + d] = b;
}

// K3: adjacency int32 -> bf16 (both orientations for WMMA-B operands)
__global__ void k_adjconv(const int* __restrict__ k_adj,
                          __bf16* __restrict__ Ab, __bf16* __restrict__ ATb) {
    int i = blockIdx.x;
    for (int j = threadIdx.x; j < KN; j += 256) {
        __bf16 v = (__bf16)(float)k_adj[(size_t)i * KN + j];
        Ab [(size_t)i * KN + j] = v;
        ATb[(size_t)j * KN + i] = v;
    }
}

// K4: stream rel_k once (b128 per lane). Per (q,k) wave:
// glob/loc logits = q_glob/q_loc . (k_enc2[k] + rel_k[q,k]).
// eg = exp(tanh(scale*glob_logit)), w = exp(tanh(scale*loc_logit)).
__global__ void k_relk(const float* __restrict__ rel_k,
                       const float* __restrict__ k_enc2,
                       const float* __restrict__ q_glob,
                       const float* __restrict__ q_loc,
                       float* __restrict__ eg, float* __restrict__ wv) {
    int wg   = blockIdx.x * 8 + (threadIdx.x >> 5);
    int lane = threadIdx.x & 31;
    int q = wg >> 9;          // 512 keys per query
    int k = wg & (KN - 1);
    const float4* rk = (const float4*)(rel_k + ((size_t)q * KN + k) * DD);
    const float4* k2 = (const float4*)(k_enc2 + (size_t)k * DD);
    const float4* qg = (const float4*)(q_glob + (size_t)q * DD);
    const float4* ql = (const float4*)(q_loc  + (size_t)q * DD);
    float dg = 0.0f, dl = 0.0f;
    #pragma unroll
    for (int i = 0; i < 2; ++i) {
        int c = lane + 32 * i;                 // 64 float4 chunks cover D=256
        float4 x = rk[c];
        float4 kk = k2[c];
        float4 g = qg[c];
        float4 l = ql[c];
        float x0 = x.x + kk.x, x1 = x.y + kk.y, x2 = x.z + kk.z, x3 = x.w + kk.w;
        dg = fmaf(g.x, x0, fmaf(g.y, x1, fmaf(g.z, x2, fmaf(g.w, x3, dg))));
        dl = fmaf(l.x, x0, fmaf(l.y, x1, fmaf(l.z, x2, fmaf(l.w, x3, dl))));
    }
    #pragma unroll
    for (int m = 16; m > 0; m >>= 1) {
        dg += __shfl_xor(dg, m, 32);
        dl += __shfl_xor(dl, m, 32);
    }
    if (lane == 0) {
        eg[(size_t)q * KN + k] = expf(tanhf(dg * SCL));
        wv[(size_t)q * KN + k] = expf(tanhf(dl * SCL));
    }
}

// K5: per-q softmax normalize eg -> glob_attn; bf16 copies of w and w*s1.
__global__ void k_rowprep(const float* __restrict__ eg, const float* __restrict__ wv,
                          const float* __restrict__ s1,
                          float* __restrict__ gattn,
                          __bf16* __restrict__ wb, __bf16* __restrict__ ws1b) {
    __shared__ float red[256];
    int q = blockIdx.x, t = threadIdx.x;
    size_t b0 = (size_t)q * KN + t, b1 = b0 + 256;
    float e0 = eg[b0], e1 = eg[b1];
    float inv = 1.0f / block_reduce_sum(e0 + e1, red);
    gattn[b0] = e0 * inv;
    gattn[b1] = e1 * inv;
    float w0 = wv[b0], w1 = wv[b1];
    wb[b0]   = (__bf16)w0;
    wb[b1]   = (__bf16)w1;
    ws1b[b0] = (__bf16)(w0 * s1[t]);
    ws1b[b1] = (__bf16)(w1 * s1[t + 256]);
}

// G1: Z = w @ A^T, T = (w*s1) @ A^T   (Q,K)x(K,K), inner j. B from Ab (row i, cols j).
__global__ void k_g1_wmma(const __bf16* __restrict__ wb, const __bf16* __restrict__ ws1b,
                          const __bf16* __restrict__ Ab,
                          float* __restrict__ Z, float* __restrict__ T) {
    int wave = blockIdx.x * (blockDim.x >> 5) + (threadIdx.x >> 5);
    int lane = threadIdx.x & 31;
    int q0 = (wave >> 5) << 4;          // 16 q-tiles
    int i0 = (wave & 31) << 4;          // 32 i-tiles
    v8f accZ = {}, accT = {};
    for (int j0 = 0; j0 < KN; j0 += 32) {
        v16bf aw = wmma_load_a(wb,   KN, q0, j0, lane);
        v16bf as = wmma_load_a(ws1b, KN, q0, j0, lane);
        v16bf b  = wmma_load_b(Ab,   KN, i0, j0, lane);  // B[kk,n]=adj[i0+n, j0+kk]
        accZ = wmma_bf16(aw, b, accZ);
        accT = wmma_bf16(as, b, accT);
    }
    wmma_store_c(Z, KN, q0, i0, lane, accZ);
    wmma_store_c(T, KN, q0, i0, lane, accT);
}

// K6: local gate; czb = glob_attn*gate/Z (bf16), r = glob_attn*(1-gate)
__global__ void k_gate(const float* __restrict__ Z, const float* __restrict__ T,
                       const float* __restrict__ s2, const float* __restrict__ blg,
                       const float* __restrict__ gattn,
                       __bf16* __restrict__ czb, float* __restrict__ rbuf) {
    int idx = blockIdx.x * 256 + threadIdx.x;
    int i = idx & (KN - 1);
    float z  = Z[idx];
    float lg = sigf(T[idx] / z + s2[i] + blg[0]);
    float ga = gattn[idx];
    czb[idx]  = (__bf16)(ga * lg / z);
    rbuf[idx] = ga * (1.0f - lg);
}

// G2: U = cz @ A   (Q,K)x(K,K), inner i. B from ATb (row j, cols i).
__global__ void k_g2_wmma(const __bf16* __restrict__ czb, const __bf16* __restrict__ ATb,
                          float* __restrict__ U) {
    int wave = blockIdx.x * (blockDim.x >> 5) + (threadIdx.x >> 5);
    int lane = threadIdx.x & 31;
    int q0 = (wave >> 5) << 4;
    int j0 = (wave & 31) << 4;
    v8f acc = {};
    for (int i0 = 0; i0 < KN; i0 += 32) {
        v16bf a = wmma_load_a(czb, KN, q0, i0, lane);
        v16bf b = wmma_load_b(ATb, KN, j0, i0, lane);   // B[kk,n]=adj[i0+kk, j0+n]
        acc = wmma_bf16(a, b, acc);
    }
    wmma_store_c(U, KN, q0, j0, lane, acc);
}

// K7: combined coefficient v = U*w + r  (bf16)
__global__ void k_comb(const float* __restrict__ U, const float* __restrict__ wv,
                       const float* __restrict__ rbuf, __bf16* __restrict__ vb) {
    int idx = blockIdx.x * 256 + threadIdx.x;
    vb[idx] = (__bf16)(fmaf(U[idx], wv[idx], rbuf[idx]));
}

// G3: k_matmul = v @ k_enc2   (Q,K)x(K,D). B from k2bT (row d, cols j).
__global__ void k_g3_wmma(const __bf16* __restrict__ vb, const __bf16* __restrict__ k2bT,
                          float* __restrict__ kmat) {
    int wave = blockIdx.x * (blockDim.x >> 5) + (threadIdx.x >> 5);
    int lane = threadIdx.x & 31;
    int q0 = (wave >> 4) << 4;          // 16 q-tiles
    int d0 = (wave & 15) << 4;          // 16 d-tiles
    v8f acc = {};
    for (int j0 = 0; j0 < KN; j0 += 32) {
        v16bf a = wmma_load_a(vb,   KN, q0, j0, lane);
        v16bf b = wmma_load_b(k2bT, KN, d0, j0, lane);  // B[kk,n]=k_enc2[j0+kk, d0+n]
        acc = wmma_bf16(a, b, acc);
    }
    wmma_store_c(kmat, DD, q0, d0, lane, acc);
}

// K8: stream rel_v once (block per q, b128 per lane), fuse final gate + output.
// Threads: u = t&63 owns float4 of D (d = 4u..4u+3); g = t>>6 splits the k-stream.
__global__ void k_final(const float* __restrict__ rel_v, const float* __restrict__ gattn,
                        const float* __restrict__ kmat, const float* __restrict__ q_enc,
                        const float* __restrict__ Wgt1, const float* __restrict__ Wgt2,
                        const float* __restrict__ bgt, float* __restrict__ out) {
    __shared__ float  gash[KN];
    __shared__ float4 part[256];
    __shared__ float  red[256];
    int q = blockIdx.x, t = threadIdx.x;
    int u = t & 63;
    int g = t >> 6;
    gash[t]       = gattn[(size_t)q * KN + t];
    gash[t + 256] = gattn[(size_t)q * KN + 256 + t];
    __syncthreads();
    const float4* rv4 = (const float4*)(rel_v + (size_t)q * KN * DD);
    float4 acc = {0.0f, 0.0f, 0.0f, 0.0f};
    for (int k = g; k < KN; k += 4) {
        float w  = gash[k];
        float4 x = rv4[(size_t)k * 64 + u];     // 32 lanes x 16B contiguous
        acc.x = fmaf(w, x.x, acc.x);
        acc.y = fmaf(w, x.y, acc.y);
        acc.z = fmaf(w, x.z, acc.z);
        acc.w = fmaf(w, x.w, acc.w);
    }
    part[t] = acc;
    __syncthreads();
    float4 qen = {0.0f, 0.0f, 0.0f, 0.0f};
    float4 qe4 = {0.0f, 0.0f, 0.0f, 0.0f};
    float partdot = 0.0f;
    if (g == 0) {
        float4 a0 = part[u], a1 = part[64 + u], a2 = part[128 + u], a3 = part[192 + u];
        float4 km = ((const float4*)(kmat + (size_t)q * DD))[u];
        qen.x = km.x + a0.x + a1.x + a2.x + a3.x;
        qen.y = km.y + a0.y + a1.y + a2.y + a3.y;
        qen.z = km.z + a0.z + a1.z + a2.z + a3.z;
        qen.w = km.w + a0.w + a1.w + a2.w + a3.w;
        qe4 = ((const float4*)(q_enc + (size_t)q * DD))[u];
        float4 w1 = ((const float4*)Wgt1)[u];
        float4 w2 = ((const float4*)Wgt2)[u];
        partdot = qen.x * w1.x + qen.y * w1.y + qen.z * w1.z + qen.w * w1.w
                + qe4.x * w2.x + qe4.y * w2.y + qe4.z * w2.z + qe4.w * w2.w;
    }
    red[t] = partdot;
    __syncthreads();
    #pragma unroll
    for (int s = 128; s > 0; s >>= 1) {
        if (t < s) red[t] += red[t + s];
        __syncthreads();
    }
    if (g == 0) {
        float gt = sigf(red[0] + bgt[0]);
        float4 h;
        h.x = gt * qen.x + (1.0f - gt) * qe4.x;
        h.y = gt * qen.y + (1.0f - gt) * qe4.y;
        h.z = gt * qen.z + (1.0f - gt) * qe4.z;
        h.w = gt * qen.w + (1.0f - gt) * qe4.w;
        ((float4*)(out + (size_t)q * DD))[u] = h;
    }
}

// ---------------------------------------------------------------------------
extern "C" void kernel_launch(void* const* d_in, const int* in_sizes, int n_in,
                              void* d_out, int out_size, void* d_ws, size_t ws_size,
                              hipStream_t stream) {
    (void)in_sizes; (void)n_in; (void)out_size; (void)ws_size;
    const float* q_enc = (const float*)d_in[0];
    const float* k_enc = (const float*)d_in[1];
    const float* rel_k = (const float*)d_in[2];
    const float* rel_v = (const float*)d_in[3];
    const int*   k_adj = (const int*)  d_in[4];
    const float* Wg  = (const float*)d_in[5];  const float* bg  = (const float*)d_in[6];
    const float* Wv0 = (const float*)d_in[7];  const float* bv0 = (const float*)d_in[8];
    const float* Wv1 = (const float*)d_in[9];  const float* bv1 = (const float*)d_in[10];
    const float* Wgq = (const float*)d_in[11]; const float* bgq = (const float*)d_in[12];
    const float* Wlq = (const float*)d_in[13]; const float* blq = (const float*)d_in[14];
    const float* Wlg1 = (const float*)d_in[15];
    const float* Wlg2 = (const float*)d_in[16];
    const float* blg  = (const float*)d_in[17];
    const float* Wgt1 = (const float*)d_in[18];
    const float* Wgt2 = (const float*)d_in[19];
    const float* bgt  = (const float*)d_in[20];
    float* out = (float*)d_out;

    char*  ws  = (char*)d_ws;
    size_t off = 0;
    auto alloc = [&](size_t bytes) -> void* {
        void* p = ws + off;
        off += (bytes + 255) & ~(size_t)255;
        return p;
    };
    float* gq     = (float*)alloc(DD * sizeof(float));
    float* g1     = (float*)alloc(DD * sizeof(float));
    float* kenc2  = (float*)alloc((size_t)KN * DD * sizeof(float));
    float* s1     = (float*)alloc(KN * sizeof(float));
    float* s2     = (float*)alloc(KN * sizeof(float));
    float* qglob  = (float*)alloc((size_t)QN * DD * sizeof(float));
    float* qloc   = (float*)alloc((size_t)QN * DD * sizeof(float));
    float* eg     = (float*)alloc((size_t)QN * KN * sizeof(float));
    float* wv     = (float*)alloc((size_t)QN * KN * sizeof(float));
    float* gattn  = (float*)alloc((size_t)QN * KN * sizeof(float));
    float* Zb     = (float*)alloc((size_t)QN * KN * sizeof(float));
    float* Tb     = (float*)alloc((size_t)QN * KN * sizeof(float));
    float* rbuf   = (float*)alloc((size_t)QN * KN * sizeof(float));
    float* Ub     = (float*)alloc((size_t)QN * KN * sizeof(float));
    float* kmat   = (float*)alloc((size_t)QN * DD * sizeof(float));
    __bf16* k2bT  = (__bf16*)alloc((size_t)DD * KN * sizeof(__bf16));
    __bf16* wb    = (__bf16*)alloc((size_t)QN * KN * sizeof(__bf16));
    __bf16* ws1b  = (__bf16*)alloc((size_t)QN * KN * sizeof(__bf16));
    __bf16* Ab    = (__bf16*)alloc((size_t)KN * KN * sizeof(__bf16));
    __bf16* ATb   = (__bf16*)alloc((size_t)KN * KN * sizeof(__bf16));
    __bf16* czb   = (__bf16*)alloc((size_t)QN * KN * sizeof(__bf16));
    __bf16* vb    = (__bf16*)alloc((size_t)QN * KN * sizeof(__bf16));

    k_prep0  <<<1,   256, 0, stream>>>(q_enc, Wg, bg, Wv0, bv0, gq, g1);
    k_adjconv<<<KN,  256, 0, stream>>>(k_adj, Ab, ATb);
    k_kenc2  <<<KN,  256, 0, stream>>>(k_enc, gq, g1, Wv1, bv1, Wlg1, Wlg2,
                                       kenc2, k2bT, s1, s2);
    k_qlin   <<<QN,  256, 0, stream>>>(q_enc, Wgq, bgq, Wlq, blq, qglob, qloc);
    k_relk   <<<(QN * KN) / 8, 256, 0, stream>>>(rel_k, kenc2, qglob, qloc, eg, wv);
    k_rowprep<<<QN,  256, 0, stream>>>(eg, wv, s1, gattn, wb, ws1b);
    k_g1_wmma<<<64,  256, 0, stream>>>(wb, ws1b, Ab, Zb, Tb);
    k_gate   <<<(QN * KN) / 256, 256, 0, stream>>>(Zb, Tb, s2, blg, gattn, czb, rbuf);
    k_g2_wmma<<<64,  256, 0, stream>>>(czb, ATb, Ub);
    k_comb   <<<(QN * KN) / 256, 256, 0, stream>>>(Ub, wv, rbuf, vb);
    k_g3_wmma<<<32,  256, 0, stream>>>(vb, k2bT, kmat);
    k_final  <<<QN,  256, 0, stream>>>(rel_v, gattn, kmat, q_enc, Wgt1, Wgt2, bgt, out);
}
